// MultiScaleAttentionModule_74646531604740
// MI455X (gfx1250) — matrix operand order repaired and checked by reading, not compiled, not run on hardware
//
#include <hip/hip_runtime.h>
#include <hip/hip_bf16.h>

#define NPIX 4096   // H*W
#define CCH  256    // C
#define CQ   32     // C/8 (attention head dim)
#define NB   4      // batch
#define VROWB 80    // padded LDS row stride (64B data + 16B pad; 20-bank lane stride)
#define NJT  (NPIX / 32)

typedef _Float16 half_t;
typedef __attribute__((ext_vector_type(16))) _Float16 v16h;
typedef __attribute__((ext_vector_type(8)))  _Float16 v8h;
typedef __attribute__((ext_vector_type(8)))  float    v8f;
typedef int v4i __attribute__((vector_size(16)));

// Async global->LDS copy path (ASYNCcnt) if the toolchain exposes it.
#if defined(__has_builtin)
#if __has_builtin(__builtin_amdgcn_global_load_async_to_lds_b128)
#define USE_ASYNC_LDS 1
#endif
#endif
#ifndef USE_ASYNC_LDS
#define USE_ASYNC_LDS 0
#endif

#if USE_ASYNC_LDS
typedef __attribute__((address_space(1))) v4i* gv4i_p;   // global v4i*
typedef __attribute__((address_space(3))) v4i* lv4i_p;   // LDS v4i*
#endif

__device__ __forceinline__ void async_wait0() {
#if USE_ASYNC_LDS
#if __has_builtin(__builtin_amdgcn_s_wait_asynccnt)
  __builtin_amdgcn_s_wait_asynccnt(0);
#else
  asm volatile("s_wait_asynccnt 0x0" ::: "memory");
#endif
#endif
}

// lane <-> lane^16 exchange (SWAPX16): xor=0x10, or=0, and=0x1f -> 0x401F
__device__ __forceinline__ float swz_xor16(float x) {
  int i = __builtin_amdgcn_ds_swizzle(__float_as_int(x), 0x401F);
  return __int_as_float(i);
}

// A-operand (16x32 f16) per-lane fragment from a 32-half contiguous row.
// lanes 0-15: K=0..15; lanes 16-31: K=16..31.
__device__ __forceinline__ v16h load_a16(const half_t* __restrict__ rowbase, bool g0) {
  const v8h lo = *(const v8h*)(rowbase + (g0 ? 0 : 16));
  const v8h hi = *(const v8h*)(rowbase + (g0 ? 8 : 24));
  v16h r;
#pragma unroll
  for (int k = 0; k < 8; ++k) { r[k] = lo[k]; r[8 + k] = hi[k]; }
  return r;
}

// Same fragment but from the padded LDS V chunk (row c, 40 halfs stride).
__device__ __forceinline__ v16h load_a16_lds(const half_t* buf, int c, bool g0) {
  const half_t* row = buf + c * (VROWB / 2);
  const v8h lo = *(const v8h*)(row + (g0 ? 0 : 16));
  const v8h hi = *(const v8h*)(row + (g0 ? 8 : 24));
  v16h r;
#pragma unroll
  for (int k = 0; k < 8; ++k) { r[k] = lo[k]; r[8 + k] = hi[k]; }
  return r;
}

// Cooperative stage of V[0..255][j0..j0+31] (16KB) into an LDS chunk.
// 1024 16B units, 128 threads x 8 units, unit u = k*128 + tid (coalesced).
__device__ __forceinline__ void stage_v(const half_t* __restrict__ vb, int j0,
                                        half_t* buf, int tid) {
#pragma unroll
  for (int k = 0; k < 8; ++k) {
    const int u = k * 128 + tid;
    const int row = u >> 2;
    const int part = u & 3;
    const half_t* g = vb + (size_t)row * NPIX + j0 + part * 8;
    half_t* l = buf + row * (VROWB / 2) + part * 8;
#if USE_ASYNC_LDS
    __builtin_amdgcn_global_load_async_to_lds_b128((gv4i_p)g, (lv4i_p)l, 0, 0);
#else
    *(v8h*)l = *(const v8h*)g;   // global_load_b128 + ds_store_b128
#endif
  }
}

// ---------------- QKV 1x1-conv projections ------------------------------------
// 16 output channels per block: each x element reused 16x (L2 traffic /16).
// qT,kT written as (B,N,32) f16 (row-contiguous); v as (B,256,N) f16.
__global__ void qkv_proj_kernel(const float* __restrict__ x,
                                const float* __restrict__ wq, const float* __restrict__ bq,
                                const float* __restrict__ wk, const float* __restrict__ bk,
                                const float* __restrict__ wv, const float* __restrict__ bv,
                                half_t* __restrict__ qT, half_t* __restrict__ kT,
                                half_t* __restrict__ vh) {
  const int n  = blockIdx.x * blockDim.x + threadIdx.x;  // pixel
  const int og = blockIdx.y * 16;                        // 0..319 in groups of 16
  const int b  = blockIdx.z;
  const float* __restrict__ xb = x + ((size_t)b * CCH) * NPIX + n;

  const float* __restrict__ w;
  const float* __restrict__ bias;
  if (og < 32)      { w = wq + (size_t)og * CCH;        bias = bq + og; }
  else if (og < 64) { w = wk + (size_t)(og - 32) * CCH; bias = bk + (og - 32); }
  else              { w = wv + (size_t)(og - 64) * CCH; bias = bv + (og - 64); }

  float acc[16];
#pragma unroll
  for (int k = 0; k < 16; ++k) acc[k] = bias[k];
#pragma unroll 4
  for (int c = 0; c < CCH; ++c) {
    const float xv = xb[(size_t)c * NPIX];
#pragma unroll
    for (int k = 0; k < 16; ++k)
      acc[k] = fmaf(w[(size_t)k * CCH + c], xv, acc[k]);  // w uniform -> scalar loads
  }

  if (og < 64) {       // q / k : one 32B vector store per thread
    v16h hv;
#pragma unroll
    for (int k = 0; k < 16; ++k) hv[k] = (half_t)acc[k];
    half_t* dst = (og < 32) ? (qT + ((size_t)b * NPIX + n) * CQ + og)
                            : (kT + ((size_t)b * NPIX + n) * CQ + (og - 32));
    *(v16h*)dst = hv;
  } else {             // v : channel-major, coalesced b16 stores
    half_t* dst = vh + ((size_t)b * CCH + (og - 64)) * NPIX + n;
#pragma unroll
    for (int k = 0; k < 16; ++k) dst[(size_t)k * NPIX] = (half_t)acc[k];
  }
}

// ---------------- fused flash attention (two-pass, transposed) ----------------
// One wave per 16-query tile, 4 waves/block sharing LDS-staged V chunks.
// S^T = K(16j x 32d) x Q^T(32d x 16i): softmax stats are per-lane scalars.
// O^T = V^T x P^T keeps 1/l scaling per-lane. Epilogue fuses gamma*(O/l)+x.
__global__ __launch_bounds__(128, 1)
void attn_kernel(const half_t* __restrict__ qT, const half_t* __restrict__ kT,
                 const half_t* __restrict__ vh, const float* __restrict__ x,
                 const float* __restrict__ gamma, float* __restrict__ out) {
  __shared__ __attribute__((aligned(16))) half_t vlds[2][CCH * (VROWB / 2)];

  const int tid  = threadIdx.x;
  const int lane = tid & 31;
  const int wave = tid >> 5;
  const int b    = blockIdx.y;
  const int i0   = (blockIdx.x * 4 + wave) * 16;
  const int l15  = lane & 15;
  const bool g0  = lane < 16;

  const half_t* __restrict__ qTb = qT + ((size_t)b * NPIX) * CQ;
  const half_t* __restrict__ kTb = kT + ((size_t)b * NPIX) * CQ;
  const half_t* __restrict__ vb  = vh + ((size_t)b * CCH) * NPIX;

  // Q^T B-operand (32x16), resident: lanes 0-15 K=0..15 of col N=lane, 16-31 K=16..31.
  const v16h Qb = *(const v16h*)(qTb + (size_t)(i0 + l15) * CQ + (g0 ? 0 : 16));

  const v8f vzero = {};

  // Kick off first V chunk; its latency hides behind all of pass 1.
  stage_v(vb, 0, vlds[0], tid);

  // ---- pass 1: streaming row max m and sum-exp l per query (per lane) ----
  float m = -__builtin_inff();
  float lsum = 0.0f;
#pragma unroll 1
  for (int j0 = 0; j0 < NPIX; j0 += 32) {
    v16h Ka0 = load_a16(kTb + (size_t)(j0 + l15) * CQ, g0);
    v16h Ka1 = load_a16(kTb + (size_t)(j0 + 16 + l15) * CQ, g0);
    v8f S0 = __builtin_amdgcn_wmma_f32_16x16x32_f16(false, Ka0, false, Qb, (short)0, vzero, false, false);
    v8f S1 = __builtin_amdgcn_wmma_f32_16x16x32_f16(false, Ka1, false, Qb, (short)0, vzero, false, false);
    float t = S0[0];
#pragma unroll
    for (int r = 1; r < 8; ++r) t = fmaxf(t, S0[r]);
#pragma unroll
    for (int r = 0; r < 8; ++r) t = fmaxf(t, S1[r]);
    t = fmaxf(t, swz_xor16(t));          // full max over this 32-key block
    const float mn   = fmaxf(m, t);
    const float corr = __expf(m - mn);   // exp(-inf)=0 on first iter
    float s = 0.0f;
#pragma unroll
    for (int r = 0; r < 8; ++r) s += __expf(S0[r] - mn) + __expf(S1[r] - mn);
    s += swz_xor16(s);                   // full sum over 32 keys
    lsum = lsum * corr + s;
    m = mn;
  }

  // ---- pass 2: O^T += V^T x P^T with fixed m (no accumulator rescale) ----
  v8f O[16];
#pragma unroll
  for (int t = 0; t < 16; ++t) O[t] = vzero;

#pragma unroll 1
  for (int jt = 0; jt < NJT; ++jt) {
    const int j0 = jt * 32;
    const half_t* cbuf = vlds[jt & 1];
    async_wait0();          // my async copies into vlds[jt&1] complete (in-order)
    __syncthreads();        // all waves' copies visible; prior readers of next buf done
    if (jt + 1 < NJT) stage_v(vb, j0 + 32, vlds[(jt + 1) & 1], tid);

    v16h Ka0 = load_a16(kTb + (size_t)(j0 + l15) * CQ, g0);
    v16h Ka1 = load_a16(kTb + (size_t)(j0 + 16 + l15) * CQ, g0);
    v8f S0 = __builtin_amdgcn_wmma_f32_16x16x32_f16(false, Ka0, false, Qb, (short)0, vzero, false, false);
    v8f S1 = __builtin_amdgcn_wmma_f32_16x16x32_f16(false, Ka1, false, Qb, (short)0, vzero, false, false);

    float e0[8], e1[8];
#pragma unroll
    for (int r = 0; r < 8; ++r) { e0[r] = __expf(S0[r] - m); e1[r] = __expf(S1[r] - m); }

    // C/D -> B layout transpose of P^T (32j x 16i) via xor-16 exchanges.
    v16h Pb;
#pragma unroll
    for (int r = 0; r < 8; ++r) {
      const float o0 = swz_xor16(e0[r]);
      const float o1 = swz_xor16(e1[r]);
      Pb[r]     = (half_t)(g0 ? e0[r] : o1);
      Pb[8 + r] = (half_t)(g0 ? o0 : e1[r]);
    }

#pragma unroll
    for (int t = 0; t < 16; ++t) {
      v16h Va = load_a16_lds(cbuf, 16 * t + l15, g0);
      O[t] = __builtin_amdgcn_wmma_f32_16x16x32_f16(false, Va, false, Pb, (short)0, O[t], false, false);
    }
  }

  // ---- epilogue: out = gamma * (O^T / l) + x ----
  const float gscale = gamma[0] * (1.0f / lsum);
  const int ii  = i0 + l15;
  const int chi = g0 ? 0 : 8;
  const size_t base = ((size_t)b * CCH) * NPIX + ii;
#pragma unroll
  for (int t = 0; t < 16; ++t) {
#pragma unroll
    for (int r = 0; r < 8; ++r) {
      const size_t idx = base + (size_t)(16 * t + chi + r) * NPIX;
      out[idx] = fmaf(gscale, O[t][r], x[idx]);
    }
  }
}

extern "C" void kernel_launch(void* const* d_in, const int* in_sizes, int n_in,
                              void* d_out, int out_size, void* d_ws, size_t ws_size,
                              hipStream_t stream) {
  (void)in_sizes; (void)n_in; (void)out_size; (void)ws_size;
  const float* x     = (const float*)d_in[0];
  const float* wq    = (const float*)d_in[1];
  const float* bq    = (const float*)d_in[2];
  const float* wk    = (const float*)d_in[3];
  const float* bk    = (const float*)d_in[4];
  const float* wv    = (const float*)d_in[5];
  const float* bv    = (const float*)d_in[6];
  const float* gamma = (const float*)d_in[7];
  float* out = (float*)d_out;

  // workspace: qT (1MB) | kT (1MB) | v (8MB), all f16
  half_t* qT = (half_t*)d_ws;
  half_t* kT = qT + (size_t)NB * NPIX * CQ;
  half_t* vh = kT + (size_t)NB * NPIX * CQ;

  dim3 gridA(NPIX / 256, (CQ + CQ + CCH) / 16, NB);   // (16, 20, 4)
  qkv_proj_kernel<<<gridA, 256, 0, stream>>>(x, wq, bq, wk, bk, wv, bv, qT, kT, vh);

  dim3 gridB(NPIX / 16 / 4, NB);                      // 4 waves/block, 1 tile/wave
  attn_kernel<<<gridB, 128, 0, stream>>>(qT, kT, vh, x, gamma, out);
}
